// CbNet_11665131176552
// MI455X (gfx1250) — compile-verified, matrix operand order checked by reading, not run on hardware
//
#include <hip/hip_runtime.h>
#include <math.h>

// ---------------- CDNA5 WMMA / vector types ----------------
typedef __attribute__((ext_vector_type(16))) _Float16 v16h;
typedef __attribute__((ext_vector_type(8)))  _Float16 v8h;
typedef __attribute__((ext_vector_type(8)))  float    v8f;

// TDM descriptor groups (per CDNA5_HIP.md probe: clang-22 = 5 args, clang-23 = 6 args)
typedef __attribute__((ext_vector_type(4))) unsigned tdm_u32x4;
typedef __attribute__((ext_vector_type(8))) int      tdm_i32x8;
typedef __attribute__((ext_vector_type(4))) int      tdm_i32x4;

#if __has_builtin(__builtin_amdgcn_tensor_load_to_lds)
#define USE_TDM 1
#endif

// ---------------- model constants ----------------
#define NRES   512
#define KNEI   64
#define SEQL   64
#define NROWS  (NRES * KNEI)     // 32768 rows for all row-wise MLPs
#define EMBED  128
#define HIDDEN 256
#define NAA2   400               // 20*20
#define PAIRD  928               // 4*128 + 400 + 16

#define ACT_NONE 0
#define ACT_RELU 1
#define ACT_GELU 2

__device__ __forceinline__ float gelu_tanh(float x) {
    float x3 = x * x * x;
    return 0.5f * x * (1.0f + tanhf(0.7978845608028654f * (x + 0.044715f * x3)));
}

// =============================================================
// Weight conversion: f32 (fi x fo, row-major) -> f16 transposed (fo x fi)
// =============================================================
__global__ void k_convT(const float* __restrict__ W, _Float16* __restrict__ Wt,
                        int fi, int fo) {
    size_t idx = (size_t)blockIdx.x * blockDim.x + threadIdx.x;
    if (idx >= (size_t)fi * fo) return;
    int c  = (int)(idx / fi);
    int kk = (int)(idx % fi);
    Wt[idx] = (_Float16)W[(size_t)kk * fo + c];
}

__global__ void k_convN(const float* __restrict__ W, _Float16* __restrict__ Wh, int count) {
    int idx = blockIdx.x * blockDim.x + threadIdx.x;
    if (idx < count) Wh[idx] = (_Float16)W[idx];
}

// =============================================================
// Generic WMMA GEMM:  Y[R x C] = act(X[R x K] @ W + bias) (+ Res)
//   X   : f32, row-major (R x K), K % 32 == 0 (true for every layer here)
//   Wt  : f16, W transposed (C x K) row-major
//   tile: 64x64, 8 waves, each wave = 16x32 slab (2 wmma / K-chunk)
//   B tile staged by the Tensor Data Mover (wave 0) when available:
//   2D tile 64 rows x 32 halves, pad_enable gives the 40-half LDS row stride.
// =============================================================
__global__ __launch_bounds__(256) void k_gemm(
    const float*    __restrict__ X,
    const _Float16* __restrict__ Wt,
    const float*    __restrict__ bias,
    float*          __restrict__ Y,
    const float*    __restrict__ Res,
    int R, int K, int C, int act)
{
    __shared__ __align__(16) _Float16 As[64][40];
    __shared__ __align__(16) _Float16 Bs[64][40];

    const int tid  = threadIdx.x;
    const int lane = tid & 31;
    const int wave = tid >> 5;                 // 0..7
    const int mw   = (wave >> 1) * 16;         // wave row base within tile
    const int nw   = (wave & 1)  * 32;         // wave col base (2 sub-tiles)
    const int rowBase = blockIdx.x * 64;
    const int colBase = blockIdx.y * 64;

    // staging coordinates: each thread moves 8 contiguous elements of one row
    const int sRow = tid >> 2;                 // 0..63
    const int sQ   = (tid & 3) * 8;            // 0,8,16,24
    const int gr   = rowBase + sRow;
    const bool rOK = gr < R;
    const float* Xrow = X + (size_t)(rOK ? gr : 0) * K + sQ;
#ifndef USE_TDM
    const int gc   = colBase + sRow;
    const bool cOK = gc < C;
    const _Float16* Wrow = Wt + (size_t)(cOK ? gc : 0) * K + sQ;
#endif

    v8f acc0 = {};
    v8f acc1 = {};
    const int fr = lane & 15;
    const int kh = (lane >> 4) * 16;

    for (int k0 = 0; k0 < K; k0 += 32) {
        // ---- stage A tile (f32 -> f16, vectorized: 2x b128 load, 1x b128 DS store)
        v8h ah = {};
        if (rOK) {
            const float4* xp = (const float4*)(Xrow + k0);
            float4 x0 = xp[0];
            float4 x1 = xp[1];
            ah[0] = (_Float16)x0.x; ah[1] = (_Float16)x0.y;
            ah[2] = (_Float16)x0.z; ah[3] = (_Float16)x0.w;
            ah[4] = (_Float16)x1.x; ah[5] = (_Float16)x1.y;
            ah[6] = (_Float16)x1.z; ah[7] = (_Float16)x1.w;
        }
        *(v8h*)&As[sRow][sQ] = ah;

#ifdef USE_TDM
        // ---- stage B tile with the Tensor Data Mover (wave 0 only; EXEC-agnostic DMA)
        if (tid < 32) {
            int cRem = C - colBase; if (cRem > 64) cRem = 64;   // OOB rows -> zeros
            unsigned long long ga =
                (unsigned long long)(const char*)(Wt + (size_t)colBase * K + k0);
            tdm_u32x4 g0;
            g0[0] = 1u;                                    // count=1 (valid descriptor)
            g0[1] = (unsigned)(size_t)(&Bs[0][0]);         // lds_addr (bytes)
            g0[2] = (unsigned)ga;                          // global_addr[31:0]
            g0[3] = ((unsigned)(ga >> 32) & 0x01FFFFFFu) | (2u << 30);  // addr[56:32], type=2
            tdm_i32x8 g1;
            // data_size=1 (2B) | pad_enable | pad_interval=3 (16 DW) | pad_amount=3 (4 DW)
            g1[0] = (int)((1u << 16) | (1u << 20) | (3u << 22) | (3u << 25));
            g1[1] = (int)(32u << 16);                      // tensor_dim0[15:0]=32 -> w1[31:16]
            g1[2] = (int)((unsigned)cRem << 16);           // dim0 hi=0 | tensor_dim1 lo
            g1[3] = (int)(32u << 16);                      // dim1 hi=0 | tile_dim0=32
            g1[4] = (int)64u;                              // tile_dim1=64 | tile_dim2=0
            g1[5] = (int)(unsigned)K;                      // tensor_dim0_stride[31:0] (halves)
            g1[6] = 0;                                     // stride hi | dim1_stride lo
            g1[7] = 0;
            tdm_i32x4 gz = {};
#if __clang_major__ >= 23
            tdm_i32x8 g4 = {};
            __builtin_amdgcn_tensor_load_to_lds(g0, g1, gz, gz, g4, 0);
#else
            __builtin_amdgcn_tensor_load_to_lds(g0, g1, gz, gz, 0);
#endif
            __builtin_amdgcn_s_wait_tensorcnt(0);
        }
#else
        // ---- fallback: vectorized b128 copy of the f16 weight tile
        v8h bh = {};
        if (cOK) bh = *(const v8h*)(Wrow + k0);
        *(v8h*)&Bs[sRow][sQ] = bh;
#endif
        __syncthreads();

        // hint the next K-tile of the activation stream (global_prefetch_b8)
        if (k0 + 32 < K) {
            __builtin_prefetch(Xrow + k0 + 32, 0, 1);
        }

        v16h a, b0, b1;
        #pragma unroll
        for (int i = 0; i < 16; ++i) a[i]  = As[mw + fr][kh + i];
        #pragma unroll
        for (int i = 0; i < 16; ++i) b0[i] = Bs[nw + fr][kh + i];
        #pragma unroll
        for (int i = 0; i < 16; ++i) b1[i] = Bs[nw + 16 + fr][kh + i];

        acc0 = __builtin_amdgcn_wmma_f32_16x16x32_f16(false, a, false, b0,
                                                      (short)0, acc0, false, false);
        acc1 = __builtin_amdgcn_wmma_f32_16x16x32_f16(false, a, false, b1,
                                                      (short)0, acc1, false, false);
        __syncthreads();
    }

    // epilogue: C/D layout -> row = r + (lane>=16)*8, col = lane&15
    const int rOff = (lane >> 4) * 8;
    const int cIn  = lane & 15;
    #pragma unroll
    for (int t = 0; t < 2; ++t) {
        v8f accv = t ? acc1 : acc0;
        int gc2 = colBase + nw + t * 16 + cIn;
        if (gc2 >= C) continue;
        #pragma unroll
        for (int r = 0; r < 8; ++r) {
            int gr2 = rowBase + mw + rOff + r;
            if (gr2 >= R) continue;
            float v = accv[r] + bias[gc2];
            if (act == ACT_RELU)      v = fmaxf(v, 0.0f);
            else if (act == ACT_GELU) v = gelu_tanh(v);
            if (Res) v += Res[(size_t)gr2 * C + gc2];
            Y[(size_t)gr2 * C + gc2] = v;
        }
    }
}

// =============================================================
// SGU spatial gating (WMMA):  out[b,t,d] = u[b,t,d] * (sum_s Wsp[t,s]*vln[b,s,d] + bsp[t])
// One block per protein b. M=t(64), N=d(128), K=s(64).
// =============================================================
__global__ __launch_bounds__(256) void k_sgu(
    const _Float16* __restrict__ WspH,   // 64x64 (t x s), f16
    const float*    __restrict__ VN,     // (NRES*64, 128) layer-normed v
    const float*    __restrict__ bsp,    // 64
    const float*    __restrict__ U,      // (NRES*64, 256), u = cols [0,128)
    float*          __restrict__ OUT)    // (NRES*64, 128)
{
    __shared__ __align__(16) _Float16 Asp[64][72];     // t x s
    __shared__ __align__(16) _Float16 Bsp[128][72];    // d x s (transposed v-stage)

    const int b    = blockIdx.x;
    const int tid  = threadIdx.x;
    const int lane = tid & 31;
    const int wave = tid >> 5;
    const int m0   = (wave >> 1) * 16;   // t base
    const int n0   = (wave & 1)  * 64;   // d base (4 sub-tiles of 16)

    // Wsp stage: 4096 halves, 16 per thread, b128 in / b128 out
    {
        int row = tid >> 2, q = (tid & 3) * 16;
        v8h h0 = *(const v8h*)(WspH + row * 64 + q);
        v8h h1 = *(const v8h*)(WspH + row * 64 + q + 8);
        *(v8h*)&Asp[row][q]     = h0;
        *(v8h*)&Asp[row][q + 8] = h1;
    }
    // v stage with transpose: float4 loads along d, b16 scatter along s
    for (int l = tid; l < 64 * 32; l += 256) {
        int s = l >> 5, dq = (l & 31) * 4;
        float4 v = *(const float4*)&VN[((size_t)b * 64 + s) * 128 + dq];
        Bsp[dq + 0][s] = (_Float16)v.x;
        Bsp[dq + 1][s] = (_Float16)v.y;
        Bsp[dq + 2][s] = (_Float16)v.z;
        Bsp[dq + 3][s] = (_Float16)v.w;
    }
    __syncthreads();

    v8f acc[4] = {};
    const int fr  = lane & 15;
    const int khB = (lane >> 4) * 16;

    #pragma unroll
    for (int kc = 0; kc < 64; kc += 32) {
        v16h a;
        #pragma unroll
        for (int i = 0; i < 16; ++i) a[i] = Asp[m0 + fr][kc + khB + i];
        #pragma unroll
        for (int t = 0; t < 4; ++t) {
            v16h bb;
            #pragma unroll
            for (int i = 0; i < 16; ++i) bb[i] = Bsp[n0 + t * 16 + fr][kc + khB + i];
            acc[t] = __builtin_amdgcn_wmma_f32_16x16x32_f16(false, a, false, bb,
                                                            (short)0, acc[t], false, false);
        }
    }

    const int rOff = (lane >> 4) * 8;
    const int cIn  = lane & 15;
    #pragma unroll
    for (int t = 0; t < 4; ++t) {
        int dcol = n0 + t * 16 + cIn;
        #pragma unroll
        for (int r = 0; r < 8; ++r) {
            int trow = m0 + rOff + r;
            float v = acc[t][r] + bsp[trow];
            float u = U[((size_t)b * 64 + trow) * 256 + dcol];
            OUT[((size_t)b * 64 + trow) * 128 + dcol] = u * v;
        }
    }
}

// =============================================================
// LayerNorm width=128, one wave32 per row (8 rows / block)
// =============================================================
__global__ __launch_bounds__(256) void k_ln(
    const float* __restrict__ X, int xStride, int xOff,
    float* __restrict__ Y,
    const float* __restrict__ g, const float* __restrict__ b, int rows)
{
    int row  = blockIdx.x * 8 + (threadIdx.x >> 5);
    int lane = threadIdx.x & 31;
    if (row >= rows) return;
    const float* xr = X + (size_t)row * xStride + xOff;
    float v0 = xr[lane], v1 = xr[lane + 32], v2 = xr[lane + 64], v3 = xr[lane + 96];
    float s = v0 + v1 + v2 + v3;
    #pragma unroll
    for (int m = 16; m; m >>= 1) s += __shfl_xor(s, m, 32);
    float mean = s * (1.0f / 128.0f);
    float d0 = v0 - mean, d1 = v1 - mean, d2 = v2 - mean, d3 = v3 - mean;
    float q = d0 * d0 + d1 * d1 + d2 * d2 + d3 * d3;
    #pragma unroll
    for (int m = 16; m; m >>= 1) q += __shfl_xor(q, m, 32);
    float inv = rsqrtf(q * (1.0f / 128.0f) + 1e-5f);
    float* yr = Y + (size_t)row * 128;
    yr[lane]      = d0 * inv * g[lane]      + b[lane];
    yr[lane + 32] = d1 * inv * g[lane + 32] + b[lane + 32];
    yr[lane + 64] = d2 * inv * g[lane + 64] + b[lane + 64];
    yr[lane + 96] = d3 * inv * g[lane + 96] + b[lane + 96];
}

// =============================================================
// Geometry features
// =============================================================
__global__ void k_crd5(const float* __restrict__ crd, float* __restrict__ crd5) {
    int i = blockIdx.x * blockDim.x + threadIdx.x;
    if (i >= NRES) return;
    const float* c = crd + (size_t)i * 12;
    float bx = c[3] - c[0],  by = c[4]  - c[1], bz = c[5]  - c[2];
    float cx = c[6] - c[3],  cy = c[7]  - c[4], cz = c[8]  - c[5];
    float ax = by * cz - bz * cy;
    float ay = bz * cx - bx * cz;
    float az = bx * cy - by * cx;
    float Cbx = -0.58273431f * ax + 0.56802827f * bx - 0.54067466f * cx + c[3];
    float Cby = -0.58273431f * ay + 0.56802827f * by - 0.54067466f * cy + c[4];
    float Cbz = -0.58273431f * az + 0.56802827f * bz - 0.54067466f * cz + c[5];
    float* o = crd5 + (size_t)i * 15;
    #pragma unroll
    for (int t = 0; t < 12; ++t) o[t] = c[t];
    o[12] = Cbx; o[13] = Cby; o[14] = Cbz;
}

// stable rank-selection == stable argsort top-64 (tie-break by index)
__global__ __launch_bounds__(128) void k_neigh(
    const float* __restrict__ crd5, int* __restrict__ idxp, float* __restrict__ dval)
{
    __shared__ float Dr[NRES];
    int i = blockIdx.x;
    float cax = crd5[(size_t)i * 15 + 3];
    float cay = crd5[(size_t)i * 15 + 4];
    float caz = crd5[(size_t)i * 15 + 5];
    for (int j = threadIdx.x; j < NRES; j += blockDim.x) {
        float dx = crd5[(size_t)j * 15 + 3] - cax;
        float dy = crd5[(size_t)j * 15 + 4] - cay;
        float dz = crd5[(size_t)j * 15 + 5] - caz;
        float D = sqrtf(dx * dx + dy * dy + dz * dz);
        Dr[j] = (j == i) ? -1.0f : D;
    }
    __syncthreads();
    for (int j = threadIdx.x; j < NRES; j += blockDim.x) {
        float dj = Dr[j];
        int rank = 0;
        for (int m = 0; m < NRES; ++m) {
            float dm = Dr[m];
            rank += (dm < dj) || (dm == dj && m < j);
        }
        if (rank < KNEI) {
            idxp[i * KNEI + rank] = j;
            dval[i * KNEI + rank] = (rank == 0) ? 0.0f : dj;
        }
    }
}

// one block per (i, neighbor): 400 rbf channels + 16 pos channels
__global__ __launch_bounds__(416) void k_rbfpos(
    const float* __restrict__ crd5, const int* __restrict__ idxp,
    float* __restrict__ rbf, float* __restrict__ pos)
{
    __shared__ float pd[25];
    int pair = blockIdx.x;
    int i = pair >> 6;
    int j = idxp[pair];
    int t = threadIdx.x;
    if (t < 25) {
        int a = t / 5, b = t % 5;
        float dx = crd5[(size_t)i * 15 + a * 3 + 0] - crd5[(size_t)j * 15 + b * 3 + 0];
        float dy = crd5[(size_t)i * 15 + a * 3 + 1] - crd5[(size_t)j * 15 + b * 3 + 1];
        float dz = crd5[(size_t)i * 15 + a * 3 + 2] - crd5[(size_t)j * 15 + b * 3 + 2];
        pd[t] = sqrtf(dx * dx + dy * dy + dz * dz);
    }
    __syncthreads();
    if (t < 400) {
        int pidx = t >> 4, c = t & 15;
        float center = (20.0f / 15.0f) * (float)c;
        float u = pd[pidx] - center;
        float v = expf(-u * u);
        if (!isfinite(v)) v = 0.0f;   // nan_to_num
        rbf[(size_t)pair * 400 + t] = v;
    } else {
        int p = t - 400;
        int f = p & 7;
        float rel  = fabsf((float)(i - j));
        float freq = expf(-logf(10000.0f) * (2.0f * (float)f) / 16.0f);
        float ang  = rel * freq;
        pos[(size_t)pair * 16 + p] = (p < 8) ? cosf(ang) : sinf(ang);
    }
}

// =============================================================
// small data movers
// =============================================================
__global__ void k_copy(const float* __restrict__ src, float* __restrict__ dst, size_t nElem) {
    size_t idx = (size_t)blockIdx.x * blockDim.x + threadIdx.x;
    if (idx < nElem) dst[idx] = src[idx];
}

__global__ __launch_bounds__(416) void k_concat2(
    const float* __restrict__ rbf, const float* __restrict__ pos, float* __restrict__ out)
{
    size_t pair = blockIdx.x;
    int t = threadIdx.x;
    out[pair * 416 + t] = (t < 400) ? rbf[pair * 400 + t] : pos[pair * 16 + (t - 400)];
}

__global__ __launch_bounds__(128) void k_gather_ftvar(
    const float* __restrict__ FT, const int* __restrict__ idxp, float* __restrict__ FTV)
{
    size_t pair = blockIdx.x;
    int j = idxp[pair];
    int c = threadIdx.x;
    FTV[pair * 128 + c] = FT[((size_t)j * SEQL + 0) * 128 + c];   // ft[j, 0, c]
}

__global__ __launch_bounds__(384) void k_concat3(
    const float* __restrict__ A, const float* __restrict__ B, const float* __restrict__ C,
    float* __restrict__ out)
{
    size_t row = blockIdx.x;
    int t = threadIdx.x;
    float v;
    if (t < 128)      v = A[row * 128 + t];
    else if (t < 256) v = B[row * 128 + (t - 128)];
    else              v = C[row * 128 + (t - 256)];
    out[row * 384 + t] = v;
}

// pair feature concat (928 wide): [ft[i,0] | ft[i,kk] | ft[j,0] | ft[j,kk] | rbf | pos]
__global__ __launch_bounds__(256) void k_outcat(
    const float* __restrict__ FT, const int* __restrict__ idxp,
    const float* __restrict__ rbf, const float* __restrict__ pos,
    float* __restrict__ out)
{
    size_t pair = blockIdx.x;
    int i  = (int)(pair >> 6);
    int kk = (int)(pair & 63);
    int j  = idxp[pair];
    for (int c = threadIdx.x; c < PAIRD; c += 256) {
        float v;
        if (c < 128)      v = FT[((size_t)i * SEQL + 0)  * 128 + c];
        else if (c < 256) v = FT[((size_t)i * SEQL + kk) * 128 + (c - 128)];
        else if (c < 384) v = FT[((size_t)j * SEQL + 0)  * 128 + (c - 256)];
        else if (c < 512) v = FT[((size_t)j * SEQL + kk) * 128 + (c - 384)];
        else if (c < 912) v = rbf[pair * 400 + (c - 512)];
        else              v = pos[pair * 16  + (c - 912)];
        out[pair * PAIRD + c] = v;
    }
}

// symmetrize: gather Calc[pi*k + row(pi,pj)], optional 20x20 transpose, mask, zero col0
__global__ __launch_bounds__(128) void k_finalize(
    const float* __restrict__ CALC, const int* __restrict__ idxp,
    const float* __restrict__ dval, const int* __restrict__ threshPtr,
    float* __restrict__ Wout, int* __restrict__ idxOut)
{
    __shared__ int rsel;
    size_t pair = blockIdx.x;
    int i  = (int)(pair >> 6);
    int kk = (int)(pair & 63);
    int j  = idxp[pair];
    int pi = min(i, j), pj = max(i, j);
    if (threadIdx.x == 0) rsel = 0;           // Nmat default 0
    __syncthreads();
    if (threadIdx.x < KNEI) {
        if (idxp[pi * KNEI + (int)threadIdx.x] == pj) rsel = threadIdx.x;
    }
    __syncthreads();
    int r = rsel;
    bool trans = i > j;
    float th = (float)(*threshPtr);
    bool keep = (dval[pair] < th) && (kk != 0);
    const float* src = CALC + ((size_t)pi * KNEI + r) * NAA2;
    for (int c = threadIdx.x; c < NAA2; c += 128) {
        int a = c / 20, b = c % 20;
        float v = trans ? src[b * 20 + a] : src[c];
        Wout[pair * NAA2 + c] = keep ? v : 0.0f;
    }
    if (threadIdx.x == 0) idxOut[pair] = j;
}

// =============================================================
// Host-side orchestration
// =============================================================
struct ResW {
    const _Float16 *WinT, *WoutT;
    const float *bin, *bout;
    const _Float16* blkW[4][2];
    const float*    blkB[4][2];
};
struct GmlpBlkW {
    const _Float16 *WinT, *WoutT, *Wsp;
    const float *bin, *bout, *bsp, *ln_b, *ln_g, *sgu_b, *sgu_g;
};
struct GmlpW {
    const _Float16* WoT;
    const float* bo;
    GmlpBlkW blk[3];
};

static void gemm(hipStream_t s, const float* X, const _Float16* Wt, const float* b,
                 float* Y, const float* Res, int R, int K, int C, int act) {
    dim3 g((R + 63) / 64, (C + 63) / 64);
    k_gemm<<<g, 256, 0, s>>>(X, Wt, b, Y, Res, R, K, C, act);
}

static void resmlp_run(hipStream_t s, const ResW& w, const float* X,
                       int fi, int fo, float* HA, float* HB, float* OUT) {
    gemm(s, X, w.WinT, w.bin, HA, nullptr, NROWS, fi, HIDDEN, ACT_RELU);
    for (int b = 0; b < 4; ++b) {
        gemm(s, HA, w.blkW[b][0], w.blkB[b][0], HB, nullptr, NROWS, HIDDEN, HIDDEN, ACT_RELU);
        // r = relu(r@W1+b1); h = h + r  -> relu then residual-add of HA, written to HA
        gemm(s, HB, w.blkW[b][1], w.blkB[b][1], HA, HA, NROWS, HIDDEN, HIDDEN, ACT_RELU);
    }
    gemm(s, HA, w.WoutT, w.bout, OUT, nullptr, NROWS, HIDDEN, fo, ACT_NONE);
}

static void gmlp_run(hipStream_t s, const GmlpW& w, float* x, float* out,
                     float* xn, float* vn, float* sg, float* y256) {
    for (int b = 0; b < 3; ++b) {
        const GmlpBlkW& bw = w.blk[b];
        k_ln<<<(NROWS + 7) / 8, 256, 0, s>>>(x, 128, 0, xn, bw.ln_g, bw.ln_b, NROWS);
        gemm(s, xn, bw.WinT, bw.bin, y256, nullptr, NROWS, 128, 256, ACT_GELU);
        k_ln<<<(NROWS + 7) / 8, 256, 0, s>>>(y256, 256, 128, vn, bw.sgu_g, bw.sgu_b, NROWS);
        k_sgu<<<NRES, 256, 0, s>>>(bw.Wsp, vn, bw.bsp, y256, sg);
        gemm(s, sg, bw.WoutT, bw.bout, x, x, NROWS, 128, 128, ACT_NONE);  // x += ...
    }
    gemm(s, x, w.WoT, w.bo, out, nullptr, NROWS, 128, 128, ACT_NONE);
}

extern "C" void kernel_launch(void* const* d_in, const int* in_sizes, int n_in,
                              void* d_out, int out_size, void* d_ws, size_t ws_size,
                              hipStream_t stream) {
    (void)in_sizes; (void)out_size; (void)ws_size;

    // ---------------- workspace bump allocator ----------------
    char* wsBase = (char*)d_ws;
    size_t wsOff = 0;
    auto alloc = [&](size_t bytes) -> void* {
        wsOff = (wsOff + 255) & ~(size_t)255;
        void* p = wsBase + wsOff;
        wsOff += bytes;
        return p;
    };
    const size_t R = NROWS;
    _Float16* wh16  = (_Float16*)alloc(3000000 * sizeof(_Float16));  // f16 weight pool
    float* crd5  = (float*)alloc(NRES * 15 * sizeof(float));
    float* dval  = (float*)alloc(R * sizeof(float));
    int*   idxp  = (int*)  alloc(R * sizeof(int));
    float* RBF   = (float*)alloc(R * 400 * sizeof(float));
    float* POS   = (float*)alloc(R * 16 * sizeof(float));
    float* BIG   = (float*)alloc(R * PAIRD * sizeof(float));  // FT0(416)/EMBIN(384)/OUT(928)
    float* FT_A  = (float*)alloc(R * 128 * sizeof(float));
    float* FT_B  = (float*)alloc(R * 128 * sizeof(float));
    float* EMB   = (float*)alloc(R * 128 * sizeof(float));
    float* INP   = (float*)alloc(R * 128 * sizeof(float));
    float* FTV   = (float*)alloc(R * 128 * sizeof(float));
    float* XN    = (float*)alloc(R * 128 * sizeof(float));
    float* VN    = (float*)alloc(R * 128 * sizeof(float));
    float* SG    = (float*)alloc(R * 128 * sizeof(float));
    float* Y256  = (float*)alloc(R * 256 * sizeof(float));
    float* HA    = (float*)alloc(R * 256 * sizeof(float));
    float* HB    = (float*)alloc(R * 256 * sizeof(float));
    float* CALC  = (float*)alloc(R * 400 * sizeof(float));

    // ---------------- weight conversion helpers ----------------
    size_t whOff = 0;
    auto convT = [&](const float* W, int fi, int fo) -> const _Float16* {
        _Float16* dst = wh16 + whOff;
        whOff += (size_t)fi * fo;
        int total = fi * fo;
        k_convT<<<(total + 255) / 256, 256, 0, stream>>>(W, dst, fi, fo);
        return dst;
    };
    auto convN = [&](const float* W, int count) -> const _Float16* {
        _Float16* dst = wh16 + whOff;
        whOff += (size_t)count;
        k_convN<<<(count + 255) / 256, 256, 0, stream>>>(W, dst, count);
        return dst;
    };

    // ---------------- parse params (jax pytree: dicts flattened by sorted key) ----
    // Top-level: crd | params | thresh.   params: gmlp, gmlp1, input_resnet,
    // input_resnet1, resnet.   gmlp: Wo, blocks[3]{Win,Wout,Wsp,bin,bout,bsp,
    // ln_b,ln_g,sgu_b,sgu_g}, bo.   resmlp: Win, Wout, bin, blocks[4][2](W,b), bout.
    const float* crd = (const float*)d_in[0];
    int p = 1;
    auto F = [&]() { return (const float*)d_in[p++]; };

    GmlpW g0, g1;
    ResW  ir, ir1, rn;
    auto parse_gmlp = [&](GmlpW& g) {
        const float* Wo = F();
        for (int b = 0; b < 3; ++b) {
            const float* Win  = F();
            const float* Wout = F();
            const float* Wsp  = F();
            g.blk[b].bin   = F();
            g.blk[b].bout  = F();
            g.blk[b].bsp   = F();
            g.blk[b].ln_b  = F();
            g.blk[b].ln_g  = F();
            g.blk[b].sgu_b = F();
            g.blk[b].sgu_g = F();
            g.blk[b].WinT  = convT(Win, 128, 256);
            g.blk[b].WoutT = convT(Wout, 128, 128);
            g.blk[b].Wsp   = convN(Wsp, 64 * 64);
        }
        g.bo  = F();
        g.WoT = convT(Wo, 128, 128);
    };
    auto parse_res = [&](ResW& r, int fi, int fo) {
        const float* Win  = F();
        const float* Wout = F();
        r.bin = F();
        for (int b = 0; b < 4; ++b)
            for (int l = 0; l < 2; ++l) {
                const float* W = F();
                r.blkB[b][l] = F();
                r.blkW[b][l] = convT(W, HIDDEN, HIDDEN);
            }
        r.bout  = F();
        r.WinT  = convT(Win, fi, HIDDEN);
        r.WoutT = convT(Wout, HIDDEN, fo);
    };
    parse_gmlp(g0);
    parse_gmlp(g1);
    parse_res(ir,  416, 128);
    parse_res(ir1, 384, 128);
    parse_res(rn,  PAIRD, NAA2);
    const int* threshPtr = (const int*)d_in[n_in - 1];

    // ---------------- features ----------------
    k_crd5<<<(NRES + 63) / 64, 64, 0, stream>>>(crd, crd5);
    k_neigh<<<NRES, 128, 0, stream>>>(crd5, idxp, dval);
    k_rbfpos<<<(int)R, 416, 0, stream>>>(crd5, idxp, RBF, POS);
    k_concat2<<<(int)R, 416, 0, stream>>>(RBF, POS, BIG);          // FT0 = [rbf|pos] (416)

    // ---------------- trunk ----------------
    resmlp_run(stream, ir, BIG, 416, 128, HA, HB, FT_A);           // ft = input_resnet(ft)
    k_copy<<<(int)((R * 128 + 255) / 256), 256, 0, stream>>>(FT_A, INP, R * 128);
    gmlp_run(stream, g0, FT_A, FT_B, XN, VN, SG, Y256);            // ft = gmlp(ft)
    float* ftcur = FT_B;
    float* ftalt = FT_A;
    for (int it = 0; it < 2; ++it) {
        k_gather_ftvar<<<(int)R, 128, 0, stream>>>(ftcur, idxp, FTV);
        k_concat3<<<(int)R, 384, 0, stream>>>(ftcur, FTV, INP, BIG);   // EMBIN (384)
        resmlp_run(stream, ir1, BIG, 384, 128, HA, HB, EMB);
        gmlp_run(stream, g1, EMB, ftalt, XN, VN, SG, Y256);
        float* t = ftcur; ftcur = ftalt; ftalt = t;
    }

    // ---------------- pair head ----------------
    k_outcat<<<(int)R, 256, 0, stream>>>(ftcur, idxp, RBF, POS, BIG);  // OUT (928)
    resmlp_run(stream, rn, BIG, PAIRD, NAA2, HA, HB, CALC);
    float* Wo = (float*)d_out;
    int* idxOut = (int*)(Wo + R * NAA2);
    k_finalize<<<(int)R, 128, 0, stream>>>(CALC, idxp, dval, threshPtr, Wo, idxOut);
}
